// MHA_46342697124027
// MI455X (gfx1250) — matrix operand order repaired and checked by reading, not compiled, not run on hardware
//
#include <hip/hip_runtime.h>
#include <hip/hip_bf16.h>

typedef _Float16 h16;
typedef __attribute__((ext_vector_type(16))) _Float16 v16h;
typedef __attribute__((ext_vector_type(8)))  _Float16 v8h;
typedef __attribute__((ext_vector_type(8)))  float    v8f;

#define D_MODEL 512
#define NHEAD   8
#define DK      64
#define LDSP    48   // padded LDS row stride (halfs) -> 96B rows, 16B-aligned chunks

// ---------------------------------------------------------------------------
// WMMA helpers (CDNA5 wave32, 16x16x32 f16 -> f32 accumulate)
// ---------------------------------------------------------------------------
__device__ __forceinline__ v8f wmma_f16(v16h a, v16h b, v8f c) {
    return __builtin_amdgcn_wmma_f32_16x16x32_f16(
        /*neg_a=*/false, a, /*neg_b=*/false, b,
        /*c_mod=*/(short)0, c, /*reuse_a=*/false, /*reuse_b=*/false);
}

// A-fragment 16x32: lane (m = lane&15) holds row m; hi = lane>>4 selects
// K-range [hi*16, hi*16+16)  (row-major source)
__device__ __forceinline__ v16h load_a_frag(const h16* __restrict__ base,
                                            int lda, int row, int k0, int hi) {
    const h16* p = base + (size_t)row * lda + k0 + hi * 16;
    v8h c0 = *(const v8h*)(p);
    v8h c1 = *(const v8h*)(p + 8);
    v16h a;
#pragma unroll
    for (int i = 0; i < 8; ++i) { a[i] = c0[i]; a[8 + i] = c1[i]; }
    return a;
}

// ---------------------------------------------------------------------------
// Kernel 1a: fp32 -> f16 elementwise convert
// ---------------------------------------------------------------------------
__global__ void k_cvt(const float* __restrict__ in, h16* __restrict__ out, int n) {
    int i = blockIdx.x * blockDim.x + threadIdx.x;
    if (i < n) out[i] = (h16)in[i];
}

// ---------------------------------------------------------------------------
// Kernel 1b: proj [H][512][64] fp32 -> projT [H][64][512] f16 (K-contiguous B)
// ---------------------------------------------------------------------------
__global__ void k_proj_t(const float* __restrict__ proj, h16* __restrict__ out) {
    int i = blockIdx.x * blockDim.x + threadIdx.x;
    if (i >= NHEAD * D_MODEL * DK) return;
    int k = i % DK;
    int d = (i / DK) % D_MODEL;
    int h = i / (DK * D_MODEL);
    out[((size_t)h * DK + k) * D_MODEL + d] = (h16)proj[i];
}

// ---------------------------------------------------------------------------
// Kernel 2: per-head projection GEMM (WMMA, K=512) + fused LayerNorm.
// One wave computes a 32-row x 64-col tile for one head.
// Ping-pong double buffering (2x unrolled K loop -> no register rotation).
// Q,K written row-major [H][n][64]; V written transposed [H][64][n].
// ---------------------------------------------------------------------------
__global__ void __launch_bounds__(32)
k_proj_ln(const h16* __restrict__ x16,     // [n][512] f16
          const h16* __restrict__ projT,   // [H][64][512] f16
          const float* __restrict__ lnw, const float* __restrict__ lnb,
          h16* __restrict__ dstQK,         // [H][n][64] or nullptr
          h16* __restrict__ dstVt,         // [H][64][n] or nullptr
          int n) {
    const int lane = threadIdx.x, m = lane & 15, hi = lane >> 4;
    const int h = blockIdx.y;
    const int row0 = blockIdx.x * 32;
    const h16* Bm = projT + (size_t)h * DK * D_MODEL;

    v8f acc[2][4] = {{{}, {}, {}, {}}, {{}, {}, {}, {}}};

    v16h a0[2], a1[2], b[2][4];

    auto load_buf = [&](int buf, int kk) {
        a0[buf] = load_a_frag(x16, D_MODEL, row0 + m, kk, hi);
        a1[buf] = load_a_frag(x16, D_MODEL, row0 + 16 + m, kk, hi);
#pragma unroll
        for (int c = 0; c < 4; ++c)
            b[buf][c] = *(const v16h*)(Bm + (size_t)(c * 16 + m) * D_MODEL + kk + hi * 16);
    };
    auto compute_buf = [&](int buf) {
#pragma unroll
        for (int c = 0; c < 4; ++c) acc[0][c] = wmma_f16(a0[buf], b[buf][c], acc[0][c]);
#pragma unroll
        for (int c = 0; c < 4; ++c) acc[1][c] = wmma_f16(a1[buf], b[buf][c], acc[1][c]);
    };

    load_buf(0, 0);
    for (int kk = 0; kk < D_MODEL; kk += 64) {   // 8 iterations, 2 K-steps each
        load_buf(1, kk + 32);                    // kk+32 <= 480 always valid
        compute_buf(0);
        load_buf(0, (kk + 64 < D_MODEL) ? kk + 64 : 0);  // clamped tail prefetch
        compute_buf(1);
    }

    float wv[4], bv[4];
#pragma unroll
    for (int c = 0; c < 4; ++c) { wv[c] = lnw[c * 16 + m]; bv[c] = lnb[c * 16 + m]; }

#pragma unroll
    for (int rg = 0; rg < 2; ++rg) {
#pragma unroll
        for (int r = 0; r < 8; ++r) {
            float s  = acc[rg][0][r] + acc[rg][1][r] + acc[rg][2][r] + acc[rg][3][r];
            float ss = acc[rg][0][r] * acc[rg][0][r] + acc[rg][1][r] * acc[rg][1][r] +
                       acc[rg][2][r] * acc[rg][2][r] + acc[rg][3][r] * acc[rg][3][r];
#pragma unroll
            for (int off = 8; off >= 1; off >>= 1) {
                s  += __shfl_xor(s,  off, 16);
                ss += __shfl_xor(ss, off, 16);
            }
            float mu  = s * (1.0f / DK);
            float var = ss * (1.0f / DK) - mu * mu;
            float rs  = rsqrtf(var + 1e-5f);
            int row = row0 + rg * 16 + r + 8 * hi;
#pragma unroll
            for (int c = 0; c < 4; ++c) {
                float y = (acc[rg][c][r] - mu) * rs * wv[c] + bv[c];
                int col = c * 16 + m;
                if (dstQK) dstQK[((size_t)h * n + row) * DK + col] = (h16)y;
                else       dstVt[((size_t)h * DK + col) * n + row] = (h16)y;
            }
        }
    }
}

// ---------------------------------------------------------------------------
// Kernel 3: fused RBF-modulated flash attention.
// One wave handles (head h, 16 query rows); keys in blocks of 32, the key
// loop is 2x unrolled with ping-pong K-fragment buffers. V fragments and
// coords/mask are issued before the softmax VALU section to hide latency.
// ---------------------------------------------------------------------------
__global__ void __launch_bounds__(32)
k_attn(const h16* __restrict__ Qh, const h16* __restrict__ Kh,
       const h16* __restrict__ Vt, const float* __restrict__ coords,
       const unsigned char* __restrict__ mask,
       h16* __restrict__ Ocat, int n) {
    __shared__ __align__(32) h16 pbuf[16 * LDSP];

    const int lane = threadIdx.x, m = lane & 15, hi = lane >> 4;
    const int h = blockIdx.y;
    const int row0 = blockIdx.x * 16;
    const h16* Qp = Qh + (size_t)h * n * DK;
    const h16* Kp = Kh + (size_t)h * n * DK;
    const h16* Vp = Vt + (size_t)h * DK * n;

    // Q A-fragments for dk 0..31 and 32..63 (reused for all key blocks)
    v16h aq0 = load_a_frag(Qp, DK, row0 + m, 0,  hi);
    v16h aq1 = load_a_frag(Qp, DK, row0 + m, 32, hi);

    float qcx[8], qcy[8], qcz[8];
#pragma unroll
    for (int r = 0; r < 8; ++r) {
        int row = row0 + r + 8 * hi;
        qcx[r] = coords[row * 3 + 0];
        qcy[r] = coords[row * 3 + 1];
        qcz[r] = coords[row * 3 + 2];
    }

    // head spread: 1 + 5*(20^(h/7)-1)/19  (torch.logspace semantics)
    float t = (float)h * (1.0f / 7.0f);
    float spread = 1.0f + 5.0f * (powf(20.0f, t) - 1.0f) * (1.0f / 19.0f);
    float nis = -0.5f / (spread * spread);
    const float scale = 0.125f;  // 1/sqrt(64)

    float mrun[8], lrun[8];
    v8f o[4] = {{}, {}, {}, {}};
#pragma unroll
    for (int r = 0; r < 8; ++r) { mrun[r] = -1e30f; lrun[r] = 0.0f; }

    // ping-pong K fragments: [buf] = {tile0 dk-lo, tile0 dk-hi, tile1 dk-lo, tile1 dk-hi}
    v16h kf[2][4];
    auto load_k = [&](int buf, int kb) {
        const int j0 = kb + m, j1 = j0 + 16;
        kf[buf][0] = *(const v16h*)(Kp + (size_t)j0 * DK + hi * 16);
        kf[buf][1] = *(const v16h*)(Kp + (size_t)j0 * DK + 32 + hi * 16);
        kf[buf][2] = *(const v16h*)(Kp + (size_t)j1 * DK + hi * 16);
        kf[buf][3] = *(const v16h*)(Kp + (size_t)j1 * DK + 32 + hi * 16);
    };

    auto process = [&](int kb, int buf) {
        const int j0 = kb + m, j1 = j0 + 16;

        // coords + mask for this block (consumed after the S-WMMAs)
        float k0x = coords[j0 * 3], k0y = coords[j0 * 3 + 1], k0z = coords[j0 * 3 + 2];
        float k1x = coords[j1 * 3], k1y = coords[j1 * 3 + 1], k1z = coords[j1 * 3 + 2];
        bool mk0 = mask[j0] != 0, mk1 = mask[j1] != 0;

        // S = Q K^T : two 16x16 tiles (keys [kb..+16) and [kb+16..+32))
        v8f s0 = {}; s0 = wmma_f16(aq0, kf[buf][0], s0); s0 = wmma_f16(aq1, kf[buf][1], s0);
        v8f s1 = {}; s1 = wmma_f16(aq0, kf[buf][2], s1); s1 = wmma_f16(aq1, kf[buf][3], s1);

        // V fragments: issue before the softmax VALU section
        v16h vf[4];
#pragma unroll
        for (int c = 0; c < 4; ++c)
            vf[c] = *(const v16h*)(Vp + (size_t)(c * 16 + m) * n + kb + hi * 16);

#pragma unroll
        for (int r = 0; r < 8; ++r) {
            float dx = qcx[r] - k0x, dy = qcy[r] - k0y, dz = qcz[r] - k0z;
            float rb0 = __expf((dx * dx + dy * dy + dz * dz) * nis);
            rb0 = fminf(fmaxf(rb0, 0.1f), 0.9f);
            dx = qcx[r] - k1x; dy = qcy[r] - k1y; dz = qcz[r] - k1z;
            float rb1 = __expf((dx * dx + dy * dy + dz * dz) * nis);
            rb1 = fminf(fmaxf(rb1, 0.1f), 0.9f);

            float sc0 = mk0 ? -1e9f : s0[r] * scale * rb0;
            float sc1 = mk1 ? -1e9f : s1[r] * scale * rb1;

            float rowm = fmaxf(sc0, sc1);
#pragma unroll
            for (int off = 8; off >= 1; off >>= 1)
                rowm = fmaxf(rowm, __shfl_xor(rowm, off, 16));
            float mn    = fmaxf(mrun[r], rowm);
            float alpha = __expf(mrun[r] - mn);
            float p0 = __expf(sc0 - mn), p1 = __expf(sc1 - mn);
            float rsum = p0 + p1;
#pragma unroll
            for (int off = 8; off >= 1; off >>= 1)
                rsum += __shfl_xor(rsum, off, 16);
            lrun[r] = lrun[r] * alpha + rsum;
            mrun[r] = mn;
#pragma unroll
            for (int c = 0; c < 4; ++c) o[c][r] *= alpha;

            // D-layout -> LDS (row r+8*hi, cols m and 16+m)
            pbuf[(r + 8 * hi) * LDSP + m]      = (h16)p0;
            pbuf[(r + 8 * hi) * LDSP + 16 + m] = (h16)p1;
        }

        // re-read P as an A-fragment (same-wave DS ops are in-order)
        v8h pc0 = *(const v8h*)(pbuf + m * LDSP + hi * 16);
        v8h pc1 = *(const v8h*)(pbuf + m * LDSP + hi * 16 + 8);
        v16h ap;
#pragma unroll
        for (int i = 0; i < 8; ++i) { ap[i] = pc0[i]; ap[8 + i] = pc1[i]; }

        // O += P * V  (V^T stored key-contiguous -> contiguous B-fragments)
#pragma unroll
        for (int c = 0; c < 4; ++c) o[c] = wmma_f16(ap, vf[c], o[c]);
    };

    load_k(0, 0);
    for (int kb = 0; kb < n; kb += 64) {           // 32 iterations, 2 blocks each
        load_k(1, kb + 32);                        // always valid (kb <= n-64)
        process(kb, 0);
        load_k(0, (kb + 64 < n) ? kb + 64 : 0);    // clamped tail prefetch
        process(kb + 32, 1);
    }

    // normalize and scatter into interleave d = k*NHEAD + h
#pragma unroll
    for (int r = 0; r < 8; ++r) {
        float inv = 1.0f / lrun[r];
        int row = row0 + r + 8 * hi;
#pragma unroll
        for (int c = 0; c < 4; ++c) {
            Ocat[(size_t)row * D_MODEL + (c * 16 + m) * NHEAD + h] = (h16)(o[c][r] * inv);
        }
    }
}

// ---------------------------------------------------------------------------
// Kernel 4: output GEMM  out[n][m] = sum_d O[n][d]*W[m][d] + bias[m]  (f32 out)
// One wave computes 32 rows x 16 cols, ping-pong double-buffered over K=512.
// ---------------------------------------------------------------------------
__global__ void __launch_bounds__(32)
k_out(const h16* __restrict__ O, const h16* __restrict__ W,
      const float* __restrict__ bias, float* __restrict__ out, int n) {
    const int lane = threadIdx.x, m = lane & 15, hi = lane >> 4;
    const int row0 = blockIdx.x * 32;
    const int col0 = blockIdx.y * 16;
    v8f acc0 = {}, acc1 = {};

    v16h a0[2], a1[2], b[2];
    auto load_buf = [&](int buf, int kk) {
        a0[buf] = load_a_frag(O, D_MODEL, row0 + m, kk, hi);
        a1[buf] = load_a_frag(O, D_MODEL, row0 + 16 + m, kk, hi);
        b[buf]  = *(const v16h*)(W + (size_t)(col0 + m) * D_MODEL + kk + hi * 16);
    };

    load_buf(0, 0);
    for (int kk = 0; kk < D_MODEL; kk += 64) {     // 8 iterations, 2 K-steps each
        load_buf(1, kk + 32);
        acc0 = wmma_f16(a0[0], b[0], acc0);
        acc1 = wmma_f16(a1[0], b[0], acc1);
        load_buf(0, (kk + 64 < D_MODEL) ? kk + 64 : 0);
        acc0 = wmma_f16(a0[1], b[1], acc0);
        acc1 = wmma_f16(a1[1], b[1], acc1);
    }

    float bb = bias[col0 + m];
#pragma unroll
    for (int r = 0; r < 8; ++r) {
        out[(size_t)(row0 + r + 8 * hi) * D_MODEL + col0 + m]      = acc0[r] + bb;
        out[(size_t)(row0 + 16 + r + 8 * hi) * D_MODEL + col0 + m] = acc1[r] + bb;
    }
}

// ---------------------------------------------------------------------------
extern "C" void kernel_launch(void* const* d_in, const int* in_sizes, int n_in,
                              void* d_out, int out_size, void* d_ws, size_t ws_size,
                              hipStream_t stream) {
    const float* q      = (const float*)d_in[0];
    const float* k      = (const float*)d_in[1];
    const float* v      = (const float*)d_in[2];
    const float* coords = (const float*)d_in[3];
    const unsigned char* mask = (const unsigned char*)d_in[4];
    const float* proj[3] = {(const float*)d_in[5], (const float*)d_in[6], (const float*)d_in[7]};
    const float* out_w  = (const float*)d_in[14];
    const float* out_b  = (const float*)d_in[15];

    const int n = in_sizes[0] / D_MODEL;      // 2048
    const size_t nel = (size_t)n * D_MODEL;

    // workspace carve-out (256B aligned)
    char* ws = (char*)d_ws;
    size_t off = 0;
    auto alloc = [&](size_t bytes) -> void* {
        void* p = ws + off;
        off = (off + bytes + 255) & ~(size_t)255;
        return p;
    };
    h16* qkv16 = (h16*)alloc(3 * nel * sizeof(h16));                       // f16 q,k,v
    h16* projT = (h16*)alloc((size_t)3 * NHEAD * DK * D_MODEL * sizeof(h16));
    h16* w16   = (h16*)alloc((size_t)D_MODEL * D_MODEL * sizeof(h16));
    h16* Qh    = (h16*)alloc((size_t)NHEAD * n * DK * sizeof(h16));
    h16* Kh    = (h16*)alloc((size_t)NHEAD * n * DK * sizeof(h16));
    h16* Vth   = (h16*)alloc((size_t)NHEAD * DK * n * sizeof(h16));
    h16* Ocat  = (h16*)alloc(nel * sizeof(h16));

    // 1) converts
    int cvt_n = (int)nel;
    k_cvt<<<(cvt_n + 255) / 256, 256, 0, stream>>>(q, qkv16, cvt_n);
    k_cvt<<<(cvt_n + 255) / 256, 256, 0, stream>>>(k, qkv16 + nel, cvt_n);
    k_cvt<<<(cvt_n + 255) / 256, 256, 0, stream>>>(v, qkv16 + 2 * nel, cvt_n);
    k_cvt<<<(D_MODEL * D_MODEL + 255) / 256, 256, 0, stream>>>(out_w, w16, D_MODEL * D_MODEL);
    for (int w = 0; w < 3; ++w)
        k_proj_t<<<(NHEAD * D_MODEL * DK + 255) / 256, 256, 0, stream>>>(
            proj[w], projT + (size_t)w * NHEAD * DK * D_MODEL);

    // 2) projection GEMMs + LayerNorm (32-row tiles)
    for (int w = 0; w < 3; ++w) {
        const float* lnw = (const float*)d_in[8 + 2 * w];
        const float* lnb = (const float*)d_in[9 + 2 * w];
        h16* dq = (w == 0) ? Qh : (w == 1) ? Kh : nullptr;
        h16* dv = (w == 2) ? Vth : nullptr;
        k_proj_ln<<<dim3(n / 32, NHEAD), 32, 0, stream>>>(
            qkv16 + (size_t)w * nel, projT + (size_t)w * NHEAD * DK * D_MODEL,
            lnw, lnb, dq, dv, n);
    }

    // 3) fused RBF flash attention
    k_attn<<<dim3(n / 16, NHEAD), 32, 0, stream>>>(Qh, Kh, Vth, coords, mask, Ocat, n);

    // 4) output projection (32-row tiles)
    k_out<<<dim3(n / 32, D_MODEL / 16), 32, 0, stream>>>(Ocat, w16, out_b, (float*)d_out, n);
}
